// QuantumAttention_46359876993379
// MI455X (gfx1250) — matrix operand order repaired and checked by reading, not compiled, not run on hardware
//
#include <hip/hip_runtime.h>

typedef unsigned short u16;
typedef __attribute__((ext_vector_type(16))) __bf16 v16bf;
typedef __attribute__((ext_vector_type(8)))  float  v8f;

union Frag {
  v16bf v;
  u16   u[16];
  uint4 q[2];
};

__device__ __forceinline__ u16 f2bf(float x) {
  unsigned u = __float_as_uint(x);
  u += 0x7FFFu + ((u >> 16) & 1u);          // round-to-nearest-even
  return (u16)(u >> 16);
}

__device__ __forceinline__ v8f wmma_bf16(const Frag& a, const Frag& b, v8f c) {
  return __builtin_amdgcn_wmma_f32_16x16x32_bf16(false, a.v, false, b.v,
                                                 (short)0, c, false, false);
}

// ---------------------------------------------------------------------------
// out = X @ W^T + bias.  X:(4096,512) f32, W:(512,512) f32 (rows of W = cols of W^T)
// mode 0: bf16 out, natural row*512+col          (Q,K projections)
// mode 1: bf16 out, per-head transposed V layout (V projections): [(b*8+h)*64+dh][l]
// mode 2: f32 out, natural                        (final Wo projection)
// Workgroup: 256 thr = 8 waves, 64x64 output tile, K-step 32 staged in LDS as bf16.
// ---------------------------------------------------------------------------
__global__ __launch_bounds__(256) void gemm_xwT_kernel(
    const float* __restrict__ X, const float* __restrict__ W,
    const float* __restrict__ bias, void* __restrict__ out, int mode)
{
  __shared__ u16 As[64][40];   // 64 rows x 32 K (+pad), bf16
  __shared__ u16 Bs[64][40];   // 64 cols x 32 K (+pad), bf16  (rows of W)
  const int tid  = threadIdx.x;
  const int wave = tid >> 5, lane = tid & 31;
  const int hi = lane >> 4, ln = lane & 15;
  const int m0g = blockIdx.x * 64, n0g = blockIdx.y * 64;
  const int wm = (wave & 3) * 16;        // wave row offset in tile
  const int wn = (wave >> 2) * 32;       // wave col offset (2 n-tiles per wave)
  const int lr = tid >> 2;               // cooperative-load row 0..63
  const int lc = (tid & 3) * 8;          // cooperative-load col base

  v8f c0 = {}, c1 = {};
  for (int k0 = 0; k0 < 512; k0 += 32) {
    const float* xp = X + (size_t)(m0g + lr) * 512 + k0 + lc;
    const float* wp = W + (size_t)(n0g + lr) * 512 + k0 + lc;
#pragma unroll
    for (int i = 0; i < 8; ++i) As[lr][lc + i] = f2bf(xp[i]);
#pragma unroll
    for (int i = 0; i < 8; ++i) Bs[lr][lc + i] = f2bf(wp[i]);
    __syncthreads();

    // A frag (16x32): lane<16 -> K 0..15 ; lane>=16 -> K 16..31 (contiguous 32B)
    Frag a, b0, b1;
    const u16* ap = &As[wm + ln][hi * 16];
    a.q[0] = *(const uint4*)ap;
    a.q[1] = *(const uint4*)(ap + 8);
    // B frag (32x16): lane<16 -> K 0..15 ; lane>=16 -> K 16..31 (contiguous 32B)
    const u16* bp0 = &Bs[wn + ln][hi * 16];
    b0.q[0] = *(const uint4*)bp0;
    b0.q[1] = *(const uint4*)(bp0 + 8);
    const u16* bp1 = &Bs[wn + 16 + ln][hi * 16];
    b1.q[0] = *(const uint4*)bp1;
    b1.q[1] = *(const uint4*)(bp1 + 8);

    c0 = wmma_bf16(a, b0, c0);
    c1 = wmma_bf16(a, b1, c1);
    __syncthreads();
  }

  // C layout: VGPR v, lanes<16 -> row v, lanes>=16 -> row v+8; col = lane&15
  const int row0 = m0g + wm + hi * 8;
  const int col0 = n0g + wn + ln;
  const int col1 = col0 + 16;
  const float bv0 = bias[col0], bv1 = bias[col1];
  if (mode == 2) {
    float* o = (float*)out;
#pragma unroll
    for (int v = 0; v < 8; ++v) {
      o[(size_t)(row0 + v) * 512 + col0] = c0[v] + bv0;
      o[(size_t)(row0 + v) * 512 + col1] = c1[v] + bv1;
    }
  } else if (mode == 0) {
    u16* o = (u16*)out;
#pragma unroll
    for (int v = 0; v < 8; ++v) {
      o[(size_t)(row0 + v) * 512 + col0] = f2bf(c0[v] + bv0);
      o[(size_t)(row0 + v) * 512 + col1] = f2bf(c1[v] + bv1);
    }
  } else {  // V transposed: [(b*8+h)*64 + dh][l]
    u16* o = (u16*)out;
    const int h0 = col0 >> 6, d0 = col0 & 63;
    const int h1 = col1 >> 6, d1 = col1 & 63;
#pragma unroll
    for (int v = 0; v < 8; ++v) {
      const int r = row0 + v;
      const int bb = r >> 11, l = r & 2047;
      o[((size_t)(bb * 8 + h0) * 64 + d0) * 2048 + l] = f2bf(c0[v] + bv0);
      o[((size_t)(bb * 8 + h1) * 64 + d1) * 2048 + l] = f2bf(c1[v] + bv1);
    }
  }
}

// ---------------------------------------------------------------------------
// Pass 1: per-row online softmax stats over s = (ir^2+ii^2)/8.
// One wave owns a 16-row i-strip of one (b,h); loops over all 128 j-tiles.
// Per-lane running (m,l); single 16-lane log-sum-exp merge at the end.
// ---------------------------------------------------------------------------
__global__ __launch_bounds__(256) void attn_stats_kernel(
    const u16* __restrict__ Qr, const u16* __restrict__ Qi,
    const u16* __restrict__ Kr, const u16* __restrict__ Ki,
    float* __restrict__ Mrow, float* __restrict__ Linv)
{
  const int lane = threadIdx.x & 31, wave = threadIdx.x >> 5;
  const int hi = lane >> 4, ln = lane & 15;
  const int bh = blockIdx.y;
  const int i0 = (blockIdx.x * 8 + wave) * 16;
  const size_t base = (size_t)(bh >> 3) * 2048 * 512 + (size_t)(bh & 7) * 64;

  Frag qr0, qr1, qi0, qi1;
  {
    const u16* p = Qr + base + (size_t)(i0 + ln) * 512 + hi * 16;
    qr0.q[0] = *(const uint4*)p;        qr0.q[1] = *(const uint4*)(p + 8);
    qr1.q[0] = *(const uint4*)(p + 32); qr1.q[1] = *(const uint4*)(p + 40);
    const u16* pi = Qi + base + (size_t)(i0 + ln) * 512 + hi * 16;
    qi0.q[0] = *(const uint4*)pi;        qi0.q[1] = *(const uint4*)(pi + 8);
    qi1.q[0] = *(const uint4*)(pi + 32); qi1.q[1] = *(const uint4*)(pi + 40);
  }

  float m[8], l[8];
#pragma unroll
  for (int v = 0; v < 8; ++v) { m[v] = -3.0e38f; l[v] = 0.0f; }

  for (int j0 = 0; j0 < 2048; j0 += 16) {
    const u16* kp = Kr + base + (size_t)(j0 + ln) * 512 + hi * 16;
    const u16* kq = Ki + base + (size_t)(j0 + ln) * 512 + hi * 16;
    __builtin_prefetch(kp + 16 * 512, 0, 0);
    __builtin_prefetch(kq + 16 * 512, 0, 0);
    Frag kr0, kr1, ki0, ki1;
    kr0.q[0] = *(const uint4*)kp;        kr0.q[1] = *(const uint4*)(kp + 8);
    kr1.q[0] = *(const uint4*)(kp + 32); kr1.q[1] = *(const uint4*)(kp + 40);
    ki0.q[0] = *(const uint4*)kq;        ki0.q[1] = *(const uint4*)(kq + 8);
    ki1.q[0] = *(const uint4*)(kq + 32); ki1.q[1] = *(const uint4*)(kq + 40);

    v8f ir = {}, t1 = {}, t2 = {};
    ir = wmma_bf16(qr0, kr0, ir); ir = wmma_bf16(qr1, kr1, ir);
    ir = wmma_bf16(qi0, ki0, ir); ir = wmma_bf16(qi1, ki1, ir);
    t1 = wmma_bf16(qr0, ki0, t1); t1 = wmma_bf16(qr1, ki1, t1);
    t2 = wmma_bf16(qi0, kr0, t2); t2 = wmma_bf16(qi1, kr1, t2);

#pragma unroll
    for (int v = 0; v < 8; ++v) {
      const float irv = ir[v], iiv = t1[v] - t2[v];
      const float s = (irv * irv + iiv * iiv) * 0.125f;
      const float mn = fmaxf(m[v], s);
      l[v] = l[v] * __expf(m[v] - mn) + __expf(s - mn);
      m[v] = mn;
    }
  }

  // merge 16 lanes of each half-wave (xor masks 1..8 stay within the half)
#pragma unroll
  for (int v = 0; v < 8; ++v) {
    float mm = m[v], ll = l[v];
#pragma unroll
    for (int mask = 1; mask < 16; mask <<= 1) {
      const float m2 = __shfl_xor(mm, mask, 32);
      const float l2 = __shfl_xor(ll, mask, 32);
      const float M = fmaxf(mm, m2);
      ll = ll * __expf(mm - M) + l2 * __expf(m2 - M);
      mm = M;
    }
    m[v] = mm; l[v] = ll;
  }
  if (ln == 0) {
#pragma unroll
    for (int v = 0; v < 8; ++v) {
      const int r = i0 + hi * 8 + v;
      Mrow[(size_t)bh * 2048 + r] = m[v];
      Linv[(size_t)bh * 2048 + r] = 1.0f / l[v];
    }
  }
}

// ---------------------------------------------------------------------------
// Pass 2: recompute ir/ii, form normalized attention + phase rotation, write
// attention to d_out, and accumulate O_r/O_i with WMMA (A = rotated probs via
// per-wave LDS C->A relayout, B = transposed V, contiguous 32B/lane).
// ---------------------------------------------------------------------------
__global__ __launch_bounds__(256) void attn_main_kernel(
    const u16* __restrict__ Qr, const u16* __restrict__ Qi,
    const u16* __restrict__ Kr, const u16* __restrict__ Ki,
    const u16* __restrict__ Vtr, const u16* __restrict__ Vti,
    const float* __restrict__ Mrow, const float* __restrict__ Linv,
    float* __restrict__ attOut, float* __restrict__ OutR, float* __restrict__ OutI)
{
  __shared__ u16 acb[8][16][40];   // per-wave 16(i) x 32(j) att*cos, bf16
  __shared__ u16 asb[8][16][40];   // per-wave att*sin
  const int lane = threadIdx.x & 31, wave = threadIdx.x >> 5;
  const int hi = lane >> 4, ln = lane & 15;
  const int bh = blockIdx.y;
  const int i0 = (blockIdx.x * 8 + wave) * 16;
  const size_t base = (size_t)(bh >> 3) * 2048 * 512 + (size_t)(bh & 7) * 64;
  const u16* vr = Vtr + (size_t)bh * 64 * 2048;
  const u16* vi = Vti + (size_t)bh * 64 * 2048;
  float* att = attOut + (size_t)bh * 2048 * 2048;

  Frag qr0, qr1, qi0, qi1;
  {
    const u16* p = Qr + base + (size_t)(i0 + ln) * 512 + hi * 16;
    qr0.q[0] = *(const uint4*)p;        qr0.q[1] = *(const uint4*)(p + 8);
    qr1.q[0] = *(const uint4*)(p + 32); qr1.q[1] = *(const uint4*)(p + 40);
    const u16* pi = Qi + base + (size_t)(i0 + ln) * 512 + hi * 16;
    qi0.q[0] = *(const uint4*)pi;        qi0.q[1] = *(const uint4*)(pi + 8);
    qi1.q[0] = *(const uint4*)(pi + 32); qi1.q[1] = *(const uint4*)(pi + 40);
  }

  float Mv[8], Lv[8];
#pragma unroll
  for (int v = 0; v < 8; ++v) {
    const int r = i0 + hi * 8 + v;
    Mv[v] = Mrow[(size_t)bh * 2048 + r];
    Lv[v] = Linv[(size_t)bh * 2048 + r];
  }

  v8f accR[4] = {};   // 16 x 64 output strip (real)
  v8f accI[4] = {};   // (imag)

  for (int j0 = 0; j0 < 2048; j0 += 32) {
#pragma unroll
    for (int sub = 0; sub < 2; ++sub) {
      const int jt = j0 + sub * 16;
      const u16* kp = Kr + base + (size_t)(jt + ln) * 512 + hi * 16;
      const u16* kq = Ki + base + (size_t)(jt + ln) * 512 + hi * 16;
      __builtin_prefetch(kp + 16 * 512, 0, 0);
      __builtin_prefetch(kq + 16 * 512, 0, 0);
      Frag kr0, kr1, ki0, ki1;
      kr0.q[0] = *(const uint4*)kp;        kr0.q[1] = *(const uint4*)(kp + 8);
      kr1.q[0] = *(const uint4*)(kp + 32); kr1.q[1] = *(const uint4*)(kp + 40);
      ki0.q[0] = *(const uint4*)kq;        ki0.q[1] = *(const uint4*)(kq + 8);
      ki1.q[0] = *(const uint4*)(kq + 32); ki1.q[1] = *(const uint4*)(kq + 40);

      v8f ir = {}, t1 = {}, t2 = {};
      ir = wmma_bf16(qr0, kr0, ir); ir = wmma_bf16(qr1, kr1, ir);
      ir = wmma_bf16(qi0, ki0, ir); ir = wmma_bf16(qi1, ki1, ir);
      t1 = wmma_bf16(qr0, ki0, t1); t1 = wmma_bf16(qr1, ki1, t1);
      t2 = wmma_bf16(qi0, kr0, t2); t2 = wmma_bf16(qi1, kr1, t2);

#pragma unroll
      for (int v = 0; v < 8; ++v) {
        const float irv = ir[v], iiv = t1[v] - t2[v];
        const float r2 = irv * irv + iiv * iiv;
        const float s = r2 * 0.125f;
        const float pw = __expf(s - Mv[v]) * Lv[v];       // normalized attention
        const bool ok = r2 > 1e-30f;
        const float rinv = ok ? rsqrtf(r2) : 0.0f;
        const float cs = ok ? irv * rinv : 1.0f;          // atan2(0,0) -> cos=1
        const float sn = iiv * rinv;
        const int rr = hi * 8 + v;
        att[(size_t)(i0 + rr) * 2048 + jt + ln] = pw;
        acb[wave][rr][sub * 16 + ln] = f2bf(pw * cs);
        asb[wave][rr][sub * 16 + ln] = f2bf(pw * sn);
      }
    }
    // per-wave private LDS tile: same-wave LDS ops are in-order; make sure
    // the C->A relayout reads see the stores
    asm volatile("s_wait_dscnt 0" ::: "memory");

    Frag aAC, aAS, aASn;
    const u16* acp = &acb[wave][ln][hi * 16];
    aAC.q[0] = *(const uint4*)acp; aAC.q[1] = *(const uint4*)(acp + 8);
    const u16* asp = &asb[wave][ln][hi * 16];
    aAS.q[0] = *(const uint4*)asp; aAS.q[1] = *(const uint4*)(asp + 8);
#pragma unroll
    for (int t = 0; t < 16; ++t) aASn.u[t] = (u16)(aAS.u[t] ^ 0x8000u);  // -as

#pragma unroll
    for (int nt = 0; nt < 4; ++nt) {
      Frag bvr, bvi;
      const u16* vp = vr + (size_t)(nt * 16 + ln) * 2048 + j0 + hi * 16;
      bvr.q[0] = *(const uint4*)vp; bvr.q[1] = *(const uint4*)(vp + 8);
      const u16* vq = vi + (size_t)(nt * 16 + ln) * 2048 + j0 + hi * 16;
      bvi.q[0] = *(const uint4*)vq; bvi.q[1] = *(const uint4*)(vq + 8);
      accR[nt] = wmma_bf16(aAC,  bvr, accR[nt]);   // + ac@Vr
      accR[nt] = wmma_bf16(aASn, bvi, accR[nt]);   // - as@Vi
      accI[nt] = wmma_bf16(aAS,  bvr, accI[nt]);   // + as@Vr
      accI[nt] = wmma_bf16(aAC,  bvi, accI[nt]);   // + ac@Vi
    }
  }

#pragma unroll
  for (int nt = 0; nt < 4; ++nt) {
#pragma unroll
    for (int v = 0; v < 8; ++v) {
      const int r = i0 + hi * 8 + v;
      const size_t idx = base + (size_t)r * 512 + nt * 16 + ln;
      OutR[idx] = accR[nt][v];
      OutI[idx] = accI[nt][v];
    }
  }
}

// ---------------------------------------------------------------------------
extern "C" void kernel_launch(void* const* d_in, const int* in_sizes, int n_in,
                              void* d_out, int out_size, void* d_ws, size_t ws_size,
                              hipStream_t stream)
{
  (void)in_sizes; (void)n_in; (void)out_size; (void)ws_size;
  const float* wave_real = (const float*)d_in[0];
  const float* wave_imag = (const float*)d_in[1];
  const float* Wq = (const float*)d_in[2];
  const float* bq = (const float*)d_in[3];
  const float* Wk = (const float*)d_in[4];
  const float* bk = (const float*)d_in[5];
  const float* Wv = (const float*)d_in[6];
  const float* bv = (const float*)d_in[7];
  const float* Wo = (const float*)d_in[8];
  const float* bo = (const float*)d_in[9];

  char* ws = (char*)d_ws;
  const size_t MB4 = (size_t)1 << 22;
  u16* Qr  = (u16*)(ws + 0 * MB4);
  u16* Qi  = (u16*)(ws + 1 * MB4);
  u16* Kr  = (u16*)(ws + 2 * MB4);
  u16* Ki  = (u16*)(ws + 3 * MB4);
  u16* Vtr = (u16*)(ws + 4 * MB4);
  u16* Vti = (u16*)(ws + 5 * MB4);
  float* Or_  = (float*)(ws + 6 * MB4);               // 8 MiB
  float* Oi_  = (float*)(ws + 8 * MB4);               // 8 MiB
  float* Mrow = (float*)(ws + 10 * MB4);              // 128 KiB
  float* Linv = (float*)(ws + 10 * MB4 + (1 << 17));  // 128 KiB

  float* outR = (float*)d_out;
  float* outI = outR + (size_t)2 * 2048 * 512;
  float* attO = outR + (size_t)4 * 2048 * 512;

  dim3 blk(256);
  dim3 gGemm(64, 8);
  gemm_xwT_kernel<<<gGemm, blk, 0, stream>>>(wave_real, Wq, bq, Qr, 0);
  gemm_xwT_kernel<<<gGemm, blk, 0, stream>>>(wave_imag, Wq, bq, Qi, 0);
  gemm_xwT_kernel<<<gGemm, blk, 0, stream>>>(wave_real, Wk, bk, Kr, 0);
  gemm_xwT_kernel<<<gGemm, blk, 0, stream>>>(wave_imag, Wk, bk, Ki, 0);
  gemm_xwT_kernel<<<gGemm, blk, 0, stream>>>(wave_real, Wv, bv, Vtr, 1);
  gemm_xwT_kernel<<<gGemm, blk, 0, stream>>>(wave_imag, Wv, bv, Vti, 1);

  dim3 gAttn(16, 16);   // x: 16 strip-groups (8 waves x 16 rows), y: b*8+h
  attn_stats_kernel<<<gAttn, blk, 0, stream>>>(Qr, Qi, Kr, Ki, Mrow, Linv);
  attn_main_kernel<<<gAttn, blk, 0, stream>>>(Qr, Qi, Kr, Ki, Vtr, Vti,
                                              Mrow, Linv, attO, Or_, Oi_);

  gemm_xwT_kernel<<<gGemm, blk, 0, stream>>>(Or_, Wo, bo, outR, 2);
  gemm_xwT_kernel<<<gGemm, blk, 0, stream>>>(Oi_, Wo, bo, outI, 2);
}
